// EdgeNetwork_23733989278143
// MI455X (gfx1250) — matrix operand order repaired and checked by reading, not compiled, not run on hardware
//
#include <hip/hip_runtime.h>
#include <hip/hip_bf16.h>

typedef __attribute__((ext_vector_type(16))) _Float16 v16h;
typedef __attribute__((ext_vector_type(8)))  _Float16 v8h;
typedef __attribute__((ext_vector_type(8)))  float    v8f;

#define N_ATOMS   50000
#define N_EDGES   400000
#define TILE_M    64                       // edges per wave-tile (4 x 16 sub-tiles)
#define N_TILES   (N_EDGES / TILE_M)       // 6250
#define NBLOCKS   392
#define WAVES_PB  8
#define TOT_WAVES (NBLOCKS * WAVES_PB)     // 3136 -> ~2 tiles/wave

__device__ __forceinline__ v8h cvt8(float4 a, float4 b) {
    v8h r;
    r[0] = (_Float16)a.x; r[1] = (_Float16)a.y; r[2] = (_Float16)a.z; r[3] = (_Float16)a.w;
    r[4] = (_Float16)b.x; r[5] = (_Float16)b.y; r[6] = (_Float16)b.z; r[7] = (_Float16)b.w;
    return r;
}

__global__ __launch_bounds__(256) void EdgeNetwork_wmma_kernel(
    const float* __restrict__ atom,    // (50000, 32)
    const float* __restrict__ bond,    // (400000, 16)
    const int*   __restrict__ pairs,   // (400000, 2) [src, dst]
    const float* __restrict__ kern,    // (16, 1024)
    const float* __restrict__ bias,    // (1024,)
    float*       __restrict__ out)     // (50000, 32), pre-zeroed
{
    // W (544x32 = 16 kernel K-blocks + bias block) pre-swizzled into the wave32
    // WMMA B-operand layout: [step s 0..16][ntile 0..1][lane 0..31][16 halves]
    __shared__ __align__(32) _Float16 kern_swz[17 * 2 * 32 * 16];     // 34816 B
    __shared__ __align__(16) _Float16 bond_lds[WAVES_PB * TILE_M * 16]; // 16 KB

    const int tid  = threadIdx.x;
    const int lane = tid & 31;
    const int wave = tid >> 5;

    // ---- cooperative fill of B operands (once per block) ----
    for (int idx = tid; idx < 17 * 2 * 32 * 16; idx += 256) {
        const int p  = idx & 15;          // half index within lane (2 per VGPR)
        const int L  = (idx >> 4) & 31;   // lane
        const int nt = (idx >> 9) & 1;    // N tile (cols 0-15 / 16-31)
        const int s  = idx >> 10;         // K step (== bond dim; 16 == bias)
        const int t  = (p & 7) + ((L >> 4) << 3) + ((p >> 3) << 4);  // K row within step (B layout)
        const int i  = nt * 16 + (L & 15);       // output column
        const float v = (s < 16) ? kern[s * 1024 + i * 32 + t] : bias[i * 32 + t];
        kern_swz[idx] = (_Float16)v;
    }
    __syncthreads();

    const v16h* kb    = (const v16h*)kern_swz;
    _Float16*   bwave = &bond_lds[wave * (TILE_M * 16)];
    const int halfsel = lane >> 4;   // 0: K-halves {0-7,16-23} / C rows 0-7
    const int ecol    = lane & 15;

    // per-wave grid-stride over 64-edge tiles (no block barriers inside)
    for (int tile = (int)blockIdx.x * WAVES_PB + wave; tile < N_TILES;
         tile += TOT_WAVES) {
        const int eb = tile * TILE_M;

        // ---- gather 4 sub-tiles of neighbor rows -> A-bases (f16, A layout) ----
        int2 pr[4];
        v16h abase[4];
        const int off = halfsel * 8;   // j-offset 0 or 8
        #pragma unroll
        for (int m = 0; m < 4; ++m) {
            pr[m] = ((const int2*)pairs)[eb + m * 16 + ecol]; // .x=src .y=dst
            const float* nrow = atom + (long)pr[m].y * 32;
            const float4 f0 = *(const float4*)(nrow + off);
            const float4 f1 = *(const float4*)(nrow + off + 4);
            const float4 f2 = *(const float4*)(nrow + off + 16);
            const float4 f3 = *(const float4*)(nrow + off + 20);
            abase[m] = __builtin_shufflevector(cvt8(f0, f1), cvt8(f2, f3),
                0, 1, 2, 3, 4, 5, 6, 7, 8, 9, 10, 11, 12, 13, 14, 15);
        }

        // ---- stage bond tile (64 edges x 16 dims) as f16 into per-wave LDS ----
        #pragma unroll
        for (int rr = 0; rr < 2; ++rr) {
            const int e2 = lane * 2 + rr;
            const float* brow = bond + (long)(eb + e2) * 16;
            const float4 g0 = *(const float4*)(brow);
            const float4 g1 = *(const float4*)(brow + 4);
            const float4 g2 = *(const float4*)(brow + 8);
            const float4 g3 = *(const float4*)(brow + 12);
            *(v8h*)&bwave[e2 * 16]     = cvt8(g0, g1);
            *(v8h*)&bwave[e2 * 16 + 8] = cvt8(g2, g3);
        }
        // same-wave DS store->load ordering + compiler barrier
        asm volatile("s_wait_dscnt 0" ::: "memory");

        v8f acc[8];
        #pragma unroll
        for (int q = 0; q < 8; ++q) acc[q] = (v8f){};

        #pragma unroll
        for (int s = 0; s < 16; ++s) {
            const v16h b0 = kb[(s * 2 + 0) * 32 + lane];
            const v16h b1 = kb[(s * 2 + 1) * 32 + lane];
            #pragma unroll
            for (int m = 0; m < 4; ++m) {
                const _Float16 c = bwave[(m * 16 + ecol) * 16 + s]; // bond[e][s]
                const v16h a = abase[m] * c;                        // 8x v_pk_mul_f16
                acc[2 * m]     = __builtin_amdgcn_wmma_f32_16x16x32_f16(
                    false, a, false, b0, (short)0, acc[2 * m],     false, false);
                acc[2 * m + 1] = __builtin_amdgcn_wmma_f32_16x16x32_f16(
                    false, a, false, b1, (short)0, acc[2 * m + 1], false, false);
            }
        }
        {   // bias step: coefficient 1 -> A = abase
            const v16h b0 = kb[(16 * 2 + 0) * 32 + lane];
            const v16h b1 = kb[(16 * 2 + 1) * 32 + lane];
            #pragma unroll
            for (int m = 0; m < 4; ++m) {
                acc[2 * m]     = __builtin_amdgcn_wmma_f32_16x16x32_f16(
                    false, abase[m], false, b0, (short)0, acc[2 * m],     false, false);
                acc[2 * m + 1] = __builtin_amdgcn_wmma_f32_16x16x32_f16(
                    false, abase[m], false, b1, (short)0, acc[2 * m + 1], false, false);
            }
        }

        // ---- scatter-add: lane holds (M = rbase+r, N = ecol) of each C tile ----
        const int rbase = halfsel * 8;
        #pragma unroll
        for (int m = 0; m < 4; ++m) {
            #pragma unroll
            for (int r = 0; r < 8; ++r) {
                const int src = __shfl(pr[m].x, rbase + r, 32);
                float* orow = out + (long)src * 32 + ecol;
                atomicAdd(orow,      acc[2 * m][r]);
                atomicAdd(orow + 16, acc[2 * m + 1][r]);
            }
        }
    }
}

__global__ void zero_f32x4(float4* p, int n4) {
    const int i = blockIdx.x * blockDim.x + threadIdx.x;
    if (i < n4) p[i] = make_float4(0.f, 0.f, 0.f, 0.f);
}

extern "C" void kernel_launch(void* const* d_in, const int* in_sizes, int n_in,
                              void* d_out, int out_size, void* d_ws, size_t ws_size,
                              hipStream_t stream) {
    const float* atom  = (const float*)d_in[0];
    const float* bond  = (const float*)d_in[1];
    const int*   pairs = (const int*)d_in[2];
    const float* kern  = (const float*)d_in[3];
    const float* bias  = (const float*)d_in[4];
    float*       out   = (float*)d_out;

    const int n4 = out_size / 4; // 1,600,000 / 4
    zero_f32x4<<<(n4 + 255) / 256, 256, 0, stream>>>((float4*)out, n4);
    EdgeNetwork_wmma_kernel<<<NBLOCKS, 256, 0, stream>>>(atom, bond, pairs, kern, bias, out);
}
